// TransformerEncoder_51599737094512
// MI455X (gfx1250) — matrix operand order repaired and checked by reading, not compiled, not run on hardware
//
#include <hip/hip_runtime.h>

// ---------------------------------------------------------------------------
// Types for CDNA5 WMMA (wave32, 16x16x32 bf16 -> f32 accumulate)
// ---------------------------------------------------------------------------
typedef __bf16 bf16;
typedef __attribute__((ext_vector_type(4)))  __bf16 v4bf;
typedef __attribute__((ext_vector_type(8)))  __bf16 v8bf;
typedef __attribute__((ext_vector_type(16))) __bf16 v16bf;
typedef __attribute__((ext_vector_type(8)))  float  v8f;
typedef __attribute__((ext_vector_type(4)))  int    v4i;

#define WMMA_BF16(a, b, c) \
  __builtin_amdgcn_wmma_f32_16x16x32_bf16(false, (a), false, (b), (short)0, (c), false, false)

// ---------------------------------------------------------------------------
// CDNA5 async LDS-DMA path (ASYNCcnt-tracked). Probed via __has_builtin so a
// toolchain without the builtins falls back to synchronous staging.
// Builtin signature (from clang diagnostic): (v4i AS1*, v4i AS3*, imm, imm).
// ---------------------------------------------------------------------------
#if defined(__HIP_DEVICE_COMPILE__) && \
    __has_builtin(__builtin_amdgcn_global_load_async_to_lds_b128) && \
    __has_builtin(__builtin_amdgcn_s_wait_asynccnt)
#define HAVE_ASYNC_LDS 1
#else
#define HAVE_ASYNC_LDS 0
#endif

#define AS1 __attribute__((address_space(1)))
#define AS3 __attribute__((address_space(3)))

#if HAVE_ASYNC_LDS
__device__ __forceinline__ void async_copy_b128(const void* gsrc, void* ldst) {
  // AS3 pointers are 32-bit; low 32 bits of a flat LDS pointer are the LDS
  // byte address (ISA 10.2).
  __builtin_amdgcn_global_load_async_to_lds_b128(
      (AS1 v4i*)(unsigned long long)gsrc,
      (AS3 v4i*)(unsigned)(unsigned long long)ldst, 0, 0);
}
__device__ __forceinline__ void async_wait0() {
  __builtin_amdgcn_s_wait_asynccnt(0);
}
#endif

// A-matrix fragment: 16x32 bf16 tile stored row-major [m][k] in LDS.
// ISA 7.12.2: lanes 0-15 hold K={0..7,16..23}, lanes 16-31 K={8..15,24..31}, m=lane%16.
__device__ __forceinline__ v16bf ld_afrag(const bf16* rp, int lane) {
  const int kb = (lane & 16) ? 8 : 0;
  v8bf lo = *(const v8bf*)(rp + kb);
  v8bf hi = *(const v8bf*)(rp + kb + 16);
  v16bf r;
#pragma unroll
  for (int i = 0; i < 8; ++i) { r[i] = lo[i]; r[i + 8] = hi[i]; }
  return r;
}

// B-matrix fragment: 32x16 tile stored transposed [n][k] in LDS (B^T rows).
// Lanes 0-15 hold K={0..7,16..23}, lanes 16-31 K={8..15,24..31}, n=lane%16.
__device__ __forceinline__ v16bf ld_bfrag(const bf16* rp, int lane) {
  const int kb = (lane & 16) ? 8 : 0;
  v8bf lo = *(const v8bf*)(rp + kb);
  v8bf hi = *(const v8bf*)(rp + kb + 16);
  v16bf r;
#pragma unroll
  for (int i = 0; i < 8; ++i) { r[i] = lo[i]; r[i + 8] = hi[i]; }
  return r;
}

__device__ __forceinline__ void vzero(v8f& v) {
#pragma unroll
  for (int i = 0; i < 8; ++i) v[i] = 0.0f;
}

// ---------------------------------------------------------------------------
// fp32 -> bf16 bulk conversion (weights staging / activation mirrors)
// ---------------------------------------------------------------------------
__global__ __launch_bounds__(256) void convert_f32_bf16(
    const float* __restrict__ src, bf16* __restrict__ dst, int n)
{
  const int i = (blockIdx.x * 256 + threadIdx.x) * 4;
  if (i + 3 < n) {
    float4 v = *(const float4*)(src + i);
    v4bf o;
    o[0] = (bf16)v.x; o[1] = (bf16)v.y; o[2] = (bf16)v.z; o[3] = (bf16)v.w;
    *(v4bf*)(dst + i) = o;
  }
}

// ---------------------------------------------------------------------------
// GEMM: Y[M,N] = act( A[M,K] @ Wt[N,K]^T + bias[N] ) (+ res[M,N])
// bf16 inputs, fp32 accumulate, fp32 or bf16 output.
// Block = 256 threads (8 waves), tile 128x128, K-step 32.
// Wave w: rows [(w>>1)*32, +32), cols [(w&1)*64, +64)  => 2x4 accum tiles.
// ---------------------------------------------------------------------------
template <bool OUT_BF16>
__global__ __launch_bounds__(256) void gemm_bf16_k(
    const bf16* __restrict__ A, const bf16* __restrict__ Wt,
    const float* __restrict__ bias, const float* __restrict__ res,
    void* __restrict__ Yv, int M, int N, int K, int relu)
{
  __shared__ __align__(16) bf16 As[128][32];
  __shared__ __align__(16) bf16 Bs[128][32];

  const int tid  = threadIdx.x;
  const int wave = tid >> 5;
  const int lane = tid & 31;
  const int wm   = (wave >> 1) * 32;   // wave row base within tile
  const int wn   = (wave & 1) * 64;    // wave col base within tile
  const int m0   = blockIdx.y * 128;
  const int n0   = blockIdx.x * 128;
  const int lr   = tid >> 1;           // 0..127 staging row
  const int lc   = (tid & 1) * 16;     // 0 or 16 staging col base

  v8f acc[2][4];
#pragma unroll
  for (int i = 0; i < 2; ++i)
#pragma unroll
    for (int j = 0; j < 4; ++j) vzero(acc[i][j]);

  const bf16* arow = A  + (size_t)(m0 + lr) * K + lc;
  const bf16* brow = Wt + (size_t)(n0 + lr) * K + lc;

  for (int k0 = 0; k0 < K; k0 += 32) {
    // Stage tiles (raw bf16 copies, 2x b128 per thread per matrix)
#if HAVE_ASYNC_LDS
    async_copy_b128(arow + k0 + 0, &As[lr][lc + 0]);
    async_copy_b128(arow + k0 + 8, &As[lr][lc + 8]);
    async_copy_b128(brow + k0 + 0, &Bs[lr][lc + 0]);
    async_copy_b128(brow + k0 + 8, &Bs[lr][lc + 8]);
#else
    *(v8bf*)&As[lr][lc + 0] = *(const v8bf*)(arow + k0 + 0);
    *(v8bf*)&As[lr][lc + 8] = *(const v8bf*)(arow + k0 + 8);
    *(v8bf*)&Bs[lr][lc + 0] = *(const v8bf*)(brow + k0 + 0);
    *(v8bf*)&Bs[lr][lc + 8] = *(const v8bf*)(brow + k0 + 8);
#endif
    if (k0 + 32 < K) {                 // warm L2/L0 for next chunk
      __builtin_prefetch(arow + k0 + 32, 0, 3);
      __builtin_prefetch(brow + k0 + 32, 0, 3);
    }
#if HAVE_ASYNC_LDS
    async_wait0();
#endif
    __syncthreads();

    v16bf a0 = ld_afrag(&As[wm +  0 + (lane & 15)][0], lane);
    v16bf a1 = ld_afrag(&As[wm + 16 + (lane & 15)][0], lane);
#pragma unroll
    for (int nt = 0; nt < 4; ++nt) {
      v16bf b = ld_bfrag(&Bs[wn + nt * 16 + (lane & 15)][0], lane);
      acc[0][nt] = WMMA_BF16(a0, b, acc[0][nt]);
      acc[1][nt] = WMMA_BF16(a1, b, acc[1][nt]);
    }
    __syncthreads();
  }

  // Epilogue. C/D layout: lane holds n=lane%16; elem v -> row v + 8*(lane/16).
  const int nl = lane & 15;
#pragma unroll
  for (int mi = 0; mi < 2; ++mi) {
    const int mlb = m0 + wm + mi * 16 + 8 * (lane >> 4);
#pragma unroll
    for (int nt = 0; nt < 4; ++nt) {
      const int n = n0 + wn + nt * 16 + nl;
      const float bv = bias ? bias[n] : 0.0f;
#pragma unroll
      for (int v = 0; v < 8; ++v) {
        const int m = mlb + v;
        float val = acc[mi][nt][v] + bv;
        if (relu) val = fmaxf(val, 0.0f);
        if (res)  val += res[(size_t)m * N + n];
        if (OUT_BF16) ((bf16*)Yv)[(size_t)m * N + n] = (bf16)val;
        else          ((float*)Yv)[(size_t)m * N + n] = val;
      }
    }
  }
}

// ---------------------------------------------------------------------------
// Flash attention (causal, GQA), bf16 in/out.
// qkv rows: [B*T, 1536] = [q(1024) | k(256) | v(256)]; head dim 64.
// grid = (T/64, B*H); block = 128 threads (4 waves).
// ---------------------------------------------------------------------------
__global__ __launch_bounds__(128) void flash_attn(
    const bf16* __restrict__ qkv, bf16* __restrict__ ctx)
{
  __shared__ __align__(16) bf16  Qs[64][64];
  __shared__ __align__(16) bf16  Ks[64][64];
  __shared__ __align__(16) bf16  Vt[64][64];   // transposed: Vt[hd][s]
  __shared__ __align__(16) float Sf[64][64];
  __shared__ __align__(16) bf16  Pb[64][64];
  __shared__ float rowm[64], rowl[64], rowa[64];

  const int tid  = threadIdx.x;
  const int wave = tid >> 5;
  const int lane = tid & 31;
  const int qi   = blockIdx.x;        // q tile 0..15
  const int bh   = blockIdx.y;        // 0..63
  const int bb   = bh >> 4;           // batch
  const int h    = bh & 15;           // q head (h = g*4 + kvh)
  const int kvh  = h & 3;
  const size_t rowbase = (size_t)bb * 1024;
  const int qcol = h * 64;
  const int kcol = 1024 + kvh * 64;
  const int vcol = 1280 + kvh * 64;

  const int lr = tid >> 1;            // 0..63
  const int lc = (tid & 1) * 32;      // 0 or 32
  const int ml = wave * 16 + 8 * (lane >> 4);
  const int nl = lane & 15;

  // Load Q tile, pre-scaled by 1/sqrt(64)
  {
    const v8bf* qp = (const v8bf*)(qkv + (rowbase + qi * 64 + lr) * 1536 + qcol + lc);
#pragma unroll
    for (int u = 0; u < 4; ++u) {
      v8bf q = qp[u];
      v8bf o;
#pragma unroll
      for (int i = 0; i < 8; ++i) o[i] = (bf16)((float)q[i] * 0.125f);
      *(v8bf*)&Qs[lr][lc + 8 * u] = o;
    }
  }
  if (tid < 64) { rowm[tid] = -3.0e38f; rowl[tid] = 0.0f; }

  v8f oacc[4];
#pragma unroll
  for (int i = 0; i < 4; ++i) vzero(oacc[i]);

  for (int j = 0; j <= qi; ++j) {
    const bf16* krow = qkv + (rowbase + j * 64 + lr) * 1536;
#pragma unroll
    for (int u = 0; u < 4; ++u) {
      // K tile: straight bf16 copy (async LDS-DMA when available)
#if HAVE_ASYNC_LDS
      async_copy_b128(krow + kcol + lc + 8 * u, &Ks[lr][lc + 8 * u]);
#else
      *(v8bf*)&Ks[lr][lc + 8 * u] = *(const v8bf*)(krow + kcol + lc + 8 * u);
#endif
      // V tile: transpose into Vt[hd][s]
      v8bf vv = *(const v8bf*)(krow + vcol + lc + 8 * u);
#pragma unroll
      for (int i = 0; i < 8; ++i) Vt[lc + 8 * u + i][lr] = vv[i];
    }
    if (j < qi) {                      // prefetch next kv tile rows
      const bf16* nrow = qkv + (rowbase + (j + 1) * 64 + lr) * 1536;
      __builtin_prefetch(nrow + kcol + lc, 0, 3);
      __builtin_prefetch(nrow + vcol + lc, 0, 3);
    }
#if HAVE_ASYNC_LDS
    async_wait0();
#endif
    __syncthreads();

    // S = Q @ K^T (wave computes 16 rows x 64 cols)
    v8f s[4];
#pragma unroll
    for (int i = 0; i < 4; ++i) vzero(s[i]);
#pragma unroll
    for (int ks = 0; ks < 64; ks += 32) {
      v16bf a = ld_afrag(&Qs[wave * 16 + (lane & 15)][ks], lane);
#pragma unroll
      for (int nt = 0; nt < 4; ++nt) {
        v16bf b = ld_bfrag(&Ks[nt * 16 + (lane & 15)][ks], lane);
        s[nt] = WMMA_BF16(a, b, s[nt]);
      }
    }
    // Causal mask on diagonal tile + spill S to LDS
#pragma unroll
    for (int nt = 0; nt < 4; ++nt) {
#pragma unroll
      for (int v = 0; v < 8; ++v) {
        float sv = s[nt][v];
        if (j == qi && (nt * 16 + nl) > (ml + v)) sv = -1.0e9f;
        Sf[ml + v][nt * 16 + nl] = sv;
      }
    }
    __syncthreads();

    // Online softmax: one thread per row
    if (tid < 64) {
      float mo = rowm[tid];
      float mx = mo;
      for (int c = 0; c < 64; ++c) mx = fmaxf(mx, Sf[tid][c]);
      const float al = __expf(mo - mx);
      float sum = 0.0f;
      for (int c = 0; c < 64; ++c) {
        float p = __expf(Sf[tid][c] - mx);
        Pb[tid][c] = (bf16)p;
        sum += p;
      }
      rowa[tid] = al;
      rowl[tid] = rowl[tid] * al + sum;
      rowm[tid] = mx;
    }
    __syncthreads();

    // Rescale running output, then O += P @ V
#pragma unroll
    for (int nt = 0; nt < 4; ++nt) {
#pragma unroll
      for (int v = 0; v < 8; ++v) oacc[nt][v] *= rowa[ml + v];
    }
#pragma unroll
    for (int ks = 0; ks < 64; ks += 32) {
      v16bf a = ld_afrag(&Pb[wave * 16 + (lane & 15)][ks], lane);
#pragma unroll
      for (int nt = 0; nt < 4; ++nt) {
        v16bf b = ld_bfrag(&Vt[nt * 16 + (lane & 15)][ks], lane);
        oacc[nt] = WMMA_BF16(a, b, oacc[nt]);
      }
    }
    __syncthreads();
  }

  // Normalize and write bf16 context (re-interleaved into [B,T,C])
#pragma unroll
  for (int v = 0; v < 8; ++v) {
    const float inv = 1.0f / rowl[ml + v];
    const size_t orow = (rowbase + qi * 64 + ml + v) * 1024;
#pragma unroll
    for (int nt = 0; nt < 4; ++nt)
      ctx[orow + h * 64 + nt * 16 + nl] = (bf16)(oacc[nt][v] * inv);
  }
}

// ---------------------------------------------------------------------------
// Row-wise LayerNorm over C=1024; fp32 out + optional bf16 mirror for next GEMM.
// ---------------------------------------------------------------------------
__global__ __launch_bounds__(256) void layernorm_k(
    const float* __restrict__ X, const float* __restrict__ g,
    const float* __restrict__ b, float* __restrict__ Y, bf16* __restrict__ Yb)
{
  __shared__ float r1[256], r2[256];
  const int row = blockIdx.x;
  const int tid = threadIdx.x;
  const float* xr = X + (size_t)row * 1024;

  float s = 0.0f, s2 = 0.0f;
  for (int c = tid; c < 1024; c += 256) {
    float v = xr[c];
    s += v; s2 += v * v;
  }
  r1[tid] = s; r2[tid] = s2;
  __syncthreads();
  for (int st = 128; st > 0; st >>= 1) {
    if (tid < st) { r1[tid] += r1[tid + st]; r2[tid] += r2[tid + st]; }
    __syncthreads();
  }
  const float mu  = r1[0] * (1.0f / 1024.0f);
  const float var = r2[0] * (1.0f / 1024.0f) - mu * mu;
  const float rs  = rsqrtf(var + 1e-5f);
  for (int c = tid; c < 1024; c += 256) {
    const float val = (xr[c] - mu) * rs * g[c] + b[c];
    Y[(size_t)row * 1024 + c] = val;
    if (Yb) Yb[(size_t)row * 1024 + c] = (bf16)val;
  }
}

// ---------------------------------------------------------------------------
// Host launcher
// ---------------------------------------------------------------------------
extern "C" void kernel_launch(void* const* d_in, const int* in_sizes, int n_in,
                              void* d_out, int out_size, void* d_ws, size_t ws_size,
                              hipStream_t stream) {
  (void)in_sizes; (void)n_in; (void)out_size; (void)ws_size;

  const float* x    = (const float*)d_in[0];
  const float* ipw  = (const float*)d_in[1];
  const float* ipb  = (const float*)d_in[2];
  const float* ow   = (const float*)d_in[3];
  const float* ob   = (const float*)d_in[4];
  const float* w1   = (const float*)d_in[5];
  const float* b1   = (const float*)d_in[6];
  const float* w2   = (const float*)d_in[7];
  const float* b2   = (const float*)d_in[8];
  const float* ln1g = (const float*)d_in[9];
  const float* ln1b = (const float*)d_in[10];
  const float* ln2g = (const float*)d_in[11];
  const float* ln2b = (const float*)d_in[12];

  const int M = 4096;                  // B*T
  const size_t MC = (size_t)M * 1024;

  // Workspace layout
  char* p = (char*)d_ws;
  float* xbuf = (float*)p;                 p += MC * 4;            // [M,1024] f32
  float* ybuf = (float*)p;                 p += MC * 4;            // [M,1024] f32
  bf16*  xb   = (bf16*)p;                  p += MC * 2;            // [M,1024] bf16
  bf16*  qkvb = (bf16*)p;                  p += (size_t)M * 1536 * 2;
  bf16*  ctxb = (bf16*)p;                  p += MC * 2;
  bf16*  ffb  = (bf16*)p;                  p += (size_t)M * 4096 * 2;
  bf16*  ipwb = (bf16*)p;                  p += (size_t)1536 * 1024 * 2;
  bf16*  owb  = (bf16*)p;                  p += (size_t)1024 * 1024 * 2;
  bf16*  w1b  = (bf16*)p;                  p += (size_t)4096 * 1024 * 2;
  bf16*  w2b  = (bf16*)p;                  p += (size_t)1024 * 4096 * 2;

  (void)hipMemcpyAsync(xbuf, x, MC * 4, hipMemcpyDeviceToDevice, stream);
  convert_f32_bf16<<<MC / 1024, 256, 0, stream>>>(x, xb, (int)MC);

  for (int l = 0; l < 4; ++l) {
    const float* ipw_l  = ipw  + (size_t)l * 1536 * 1024;
    const float* ipb_l  = ipb  + (size_t)l * 1536;
    const float* ow_l   = ow   + (size_t)l * 1024 * 1024;
    const float* ob_l   = ob   + (size_t)l * 1024;
    const float* w1_l   = w1   + (size_t)l * 4096 * 1024;
    const float* b1_l   = b1   + (size_t)l * 4096;
    const float* w2_l   = w2   + (size_t)l * 1024 * 4096;
    const float* b2_l   = b2   + (size_t)l * 1024;

    // Stage this layer's weights as bf16
    convert_f32_bf16<<<1536 * 1024 / 1024, 256, 0, stream>>>(ipw_l, ipwb, 1536 * 1024);
    convert_f32_bf16<<<1024 * 1024 / 1024, 256, 0, stream>>>(ow_l,  owb,  1024 * 1024);
    convert_f32_bf16<<<4096 * 1024 / 1024, 256, 0, stream>>>(w1_l,  w1b,  4096 * 1024);
    convert_f32_bf16<<<1024 * 4096 / 1024, 256, 0, stream>>>(w2_l,  w2b,  1024 * 4096);

    // qkv = x @ ipw^T + ipb  (bf16 out, consumed by attention)
    gemm_bf16_k<true><<<dim3(1536 / 128, M / 128), 256, 0, stream>>>(
        xb, ipwb, ipb_l, nullptr, qkvb, M, 1536, 1024, 0);

    // causal GQA flash attention -> ctx (bf16)
    flash_attn<<<dim3(16, 64), 128, 0, stream>>>(qkvb, ctxb);

    // attn_out = ctx @ ow^T + ob + x  (fp32 out, residual fused)
    gemm_bf16_k<false><<<dim3(1024 / 128, M / 128), 256, 0, stream>>>(
        ctxb, owb, ob_l, xbuf, ybuf, M, 1024, 1024, 0);

    // x = LN1(attn_out): fp32 + bf16 mirror
    layernorm_k<<<M, 256, 0, stream>>>(ybuf, ln1g + (size_t)l * 1024,
                                       ln1b + (size_t)l * 1024, xbuf, xb);

    // ff = relu(x @ w1^T + b1)  (bf16 out)
    gemm_bf16_k<true><<<dim3(4096 / 128, M / 128), 256, 0, stream>>>(
        xb, w1b, b1_l, nullptr, ffb, M, 4096, 1024, 1);

    // y = ff @ w2^T + b2 + x  (fp32 out, residual fused)
    gemm_bf16_k<false><<<dim3(1024 / 128, M / 128), 256, 0, stream>>>(
        ffb, w2b, b2_l, xbuf, ybuf, M, 1024, 4096, 0);

    // x = LN2(y); final layer writes fp32 d_out only
    float* lnout = (l == 3) ? (float*)d_out : xbuf;
    bf16*  lnoutb = (l == 3) ? nullptr : xb;
    layernorm_k<<<M, 256, 0, stream>>>(ybuf, ln2g + (size_t)l * 1024,
                                       ln2b + (size_t)l * 1024, lnout, lnoutb);
  }
}
